// ShmmLm_75153337745780
// MI455X (gfx1250) — matrix operand order repaired and checked by reading, not compiled, not run on hardware
//
#include <hip/hip_runtime.h>
#include <hip/hip_bf16.h>

#define V_   16000
#define C_   4096
#define H_   256
#define SPW_ 32
#define N_   32
#define T_   128

typedef __attribute__((ext_vector_type(16))) _Float16 v16h;
typedef __attribute__((ext_vector_type(8)))  _Float16 v8h;
typedef __attribute__((ext_vector_type(8)))  float    v8f;

__device__ __forceinline__ v8f wmma16(v16h a, v16h b, v8f c) {
  // D = A(16x32 f16) x B(32x16 f16) + C(16x16 f32)
  return __builtin_amdgcn_wmma_f32_16x16x32_f16(false, a, false, b, (short)0, c, false, false);
}

__device__ __forceinline__ v16h frag16(v8h lo, v8h hi2) {
  return __builtin_shufflevector(lo, hi2, 0, 1, 2, 3, 4, 5, 6, 7,
                                 8, 9, 10, 11, 12, 13, 14, 15);
}

// ---- ordered-uint encoding for float atomicMax ----
__device__ __forceinline__ unsigned fenc(float f) {
  unsigned u = __float_as_uint(f);
  return (u & 0x80000000u) ? ~u : (u | 0x80000000u);
}
__device__ __forceinline__ float fdec(unsigned k) {
  unsigned u = (k & 0x80000000u) ? (k ^ 0x80000000u) : ~k;
  return __uint_as_float(u);
}

// ============================================================
// Weight prep: W (H_ x H_, f32, k-major) -> Wh (n-major f16) so that
// each lane's B fragment is two 16-byte runs from one Wh row.
// ============================================================
__global__ void wtrans_f16_kernel(const float* __restrict__ W, _Float16* __restrict__ Wh)
{
  __shared__ float tile[32][33];
  int k0 = blockIdx.x * 32, n0 = blockIdx.y * 32;
  for (int r = threadIdx.y; r < 32; r += 8)
    tile[r][threadIdx.x] = W[(size_t)(k0 + r) * H_ + n0 + threadIdx.x];
  __syncthreads();
  for (int r = threadIdx.y; r < 32; r += 8)
    Wh[(size_t)(n0 + r) * H_ + k0 + threadIdx.x] = (_Float16)tile[threadIdx.x][r];
}

// elementwise f32 -> f16 (for next_state_proj, already row-per-lane friendly)
__global__ void cvt_f16_kernel(const float* __restrict__ src, _Float16* __restrict__ dst, int n)
{
  int idx = blockIdx.x * 256 + threadIdx.x;
  if (idx < n) dst[idx] = (_Float16)src[idx];
}

// ============================================================
// Residual MLP: Y = relu(relu(X@W1+b1)@W2+b2) + X   (X: C_ x H_)
// W1h/W2h are n-major f16. One block = 16 rows; 16 waves = 16 col tiles.
// ============================================================
__global__ __launch_bounds__(512) void mlp_residual_kernel(
    const float* __restrict__ X, const _Float16* __restrict__ W1h, const float* __restrict__ b1,
    const _Float16* __restrict__ W2h, const float* __restrict__ b2, float* __restrict__ Y)
{
  __shared__ alignas(16) _Float16 Xh[16][H_ + 8];
  __shared__ alignas(16) _Float16 Hh[16][H_ + 8];
  const int tid = threadIdx.x, wav = tid >> 5, lane = tid & 31;
  const int m0 = blockIdx.x * 16;
  for (int idx = tid; idx < 16 * H_; idx += 512) {
    int r = idx >> 8, cc = idx & (H_ - 1);
    Xh[r][cc] = (_Float16)X[(size_t)(m0 + r) * H_ + cc];
  }
  __syncthreads();
  const int n0 = wav * 16;
  const int mrow = lane & 15;
  const int hi = lane >> 4;   // half-wave selector

  v8f acc = {};
  {
    const _Float16* brow = W1h + (size_t)(n0 + mrow) * H_ + hi * 8;
    for (int k0 = 0; k0 < H_; k0 += 32) {
      v16h a = frag16(*(const v8h*)&Xh[mrow][k0 + hi * 8],
                      *(const v8h*)&Xh[mrow][k0 + 16 + hi * 8]);
      const v8h* bp = (const v8h*)(brow + k0);
      v16h b = frag16(bp[0], bp[2]);
      acc = wmma16(a, b, acc);
    }
  }
#pragma unroll
  for (int g = 0; g < 8; ++g) {
    int m = g + hi * 8, n = n0 + mrow;
    float v = acc[g] + b1[n];
    Hh[m][n] = (_Float16)(v > 0.f ? v : 0.f);
  }
  __syncthreads();
  v8f acc2 = {};
  {
    const _Float16* brow = W2h + (size_t)(n0 + mrow) * H_ + hi * 8;
    for (int k0 = 0; k0 < H_; k0 += 32) {
      v16h a = frag16(*(const v8h*)&Hh[mrow][k0 + hi * 8],
                      *(const v8h*)&Hh[mrow][k0 + 16 + hi * 8]);
      const v8h* bp = (const v8h*)(brow + k0);
      v16h b = frag16(bp[0], bp[2]);
      acc2 = wmma16(a, b, acc2);
    }
  }
#pragma unroll
  for (int g = 0; g < 8; ++g) {
    int m = g + hi * 8, n = n0 + mrow;
    float v = acc2[g] + b2[n];
    v = v > 0.f ? v : 0.f;
    Y[(size_t)(m0 + m) * H_ + n] = v + X[(size_t)(m0 + m) * H_ + n];
  }
}

// ============================================================
// trans_logits[i,j] = Yt[i,:] . nsp[j,:]   (C_ x C_), nsph = f16 copy of nsp
// grid (C_/16 rows, C_/256 col-groups), 16 waves per block.
// ============================================================
__global__ __launch_bounds__(512) void trans_gemm_kernel(
    const float* __restrict__ A, const _Float16* __restrict__ Ph, float* __restrict__ out)
{
  __shared__ alignas(16) _Float16 Ah[16][H_ + 8];
  const int tid = threadIdx.x, wav = tid >> 5, lane = tid & 31;
  const int i0 = blockIdx.x * 16;
  const int j0 = blockIdx.y * 256 + wav * 16;
  for (int idx = tid; idx < 16 * H_; idx += 512) {
    int r = idx >> 8, cc = idx & (H_ - 1);
    Ah[r][cc] = (_Float16)A[(size_t)(i0 + r) * H_ + cc];
  }
  __syncthreads();
  const int mrow = lane & 15, hi = lane >> 4;
  const _Float16* brow = Ph + (size_t)(j0 + mrow) * H_ + hi * 8;
  v8f acc = {};
  for (int k0 = 0; k0 < H_; k0 += 32) {
    v16h a = frag16(*(const v8h*)&Ah[mrow][k0 + hi * 8],
                    *(const v8h*)&Ah[mrow][k0 + 16 + hi * 8]);
    const v8h* bp = (const v8h*)(brow + k0);
    v16h b = frag16(bp[0], bp[2]);
    acc = wmma16(a, b, acc);
  }
#pragma unroll
  for (int g = 0; g < 8; ++g) {
    int m = g + hi * 8;
    out[(size_t)(i0 + m) * C_ + (j0 + mrow)] = acc[g];
  }
}

// ============================================================
// s_logits[c] = Ys[c,:] . s_out_w + s_out_b ; one wave per row
// ============================================================
__global__ __launch_bounds__(256) void sout_kernel(
    const float* __restrict__ Ys, const float* __restrict__ w,
    const float* __restrict__ b, float* __restrict__ slog)
{
  int r = blockIdx.x * 8 + (threadIdx.x >> 5);
  int lane = threadIdx.x & 31;
  float s = 0.f;
  for (int h = lane; h < H_; h += 32) s += Ys[(size_t)r * H_ + h] * w[h];
  for (int o = 16; o; o >>= 1) s += __shfl_xor(s, o);
  if (lane == 0) slog[r] = s + b[0];
}

// ============================================================
// In-place row-wise log-softmax (one block per row)
// ============================================================
__global__ __launch_bounds__(256) void row_logsoftmax_kernel(float* __restrict__ data, int width)
{
  float* row = data + (size_t)blockIdx.x * width;
  __shared__ float red[256];
  int tid = threadIdx.x;
  float m = -__builtin_inff();
  for (int c = tid; c < width; c += 256) m = fmaxf(m, row[c]);
  red[tid] = m; __syncthreads();
  for (int s = 128; s; s >>= 1) { if (tid < s) red[tid] = fmaxf(red[tid], red[tid + s]); __syncthreads(); }
  m = red[0]; __syncthreads();
  float su = 0.f;
  for (int c = tid; c < width; c += 256) su += expf(row[c] - m);
  red[tid] = su; __syncthreads();
  for (int s = 128; s; s >>= 1) { if (tid < s) red[tid] += red[tid + s]; __syncthreads(); }
  float l = m + logf(red[0]);
  for (int c = tid; c < width; c += 256) row[c] -= l;
}

// ============================================================
// Transpose e_out_w (H_ x V_) -> Wt (V_ x H_)
// ============================================================
__global__ void transpose_kernel(const float* __restrict__ W, float* __restrict__ Wt)
{
  __shared__ float tile[32][33];
  int v0 = blockIdx.x * 32, h0 = blockIdx.y * 32;
  for (int r = threadIdx.y; r < 32; r += 8)
    tile[r][threadIdx.x] = W[(size_t)(h0 + r) * V_ + v0 + threadIdx.x];
  __syncthreads();
  for (int r = threadIdx.y; r < 32; r += 8)
    Wt[(size_t)(v0 + r) * H_ + h0 + threadIdx.x] = tile[threadIdx.x][r];
}

__global__ void init_kernel(unsigned* __restrict__ maxkey, float* __restrict__ ssum)
{
  int c = blockIdx.x * 256 + threadIdx.x;
  if (c < C_) { maxkey[c] = 0x007FFFFFu; /* fenc(-inf) */ ssum[c] = 0.f; }
}

// ============================================================
// Emission sparse dots: one wave per word v, lane k handles state w2s[v,k].
// Phase 1: D[v,k] = Ye[c,:].Wt[v,:] + eb[v]; atomicMax per-state (dedup per word)
// ============================================================
__global__ __launch_bounds__(256) void emis_dot_kernel(
    const float* __restrict__ Ye, const float* __restrict__ Wt, const float* __restrict__ eb,
    const int* __restrict__ w2s, float* __restrict__ D, unsigned* __restrict__ maxkey)
{
  int v = blockIdx.x * 8 + (threadIdx.x >> 5);
  int k = threadIdx.x & 31;
  int c = w2s[(size_t)v * SPW_ + k];
  const float* wr = Wt + (size_t)v * H_;
  const float* yr = Ye + (size_t)c * H_;
  float s = 0.f;
  for (int h = 0; h < H_; ++h) s += yr[h] * wr[h];
  s += eb[v];
  D[(size_t)v * SPW_ + k] = s;
  bool dup = false;
  for (int j = 0; j < 31; ++j) { int cj = __shfl(c, j); if (j < k && cj == c) dup = true; }
  if (!dup) atomicMax(&maxkey[c], fenc(s));
}

// Phase 2: per-state sum of exp (deduped)
__global__ __launch_bounds__(256) void emis_sum_kernel(
    const float* __restrict__ D, const int* __restrict__ w2s,
    const unsigned* __restrict__ maxkey, float* __restrict__ ssum)
{
  int v = blockIdx.x * 8 + (threadIdx.x >> 5);
  int k = threadIdx.x & 31;
  int c = w2s[(size_t)v * SPW_ + k];
  float s = D[(size_t)v * SPW_ + k];
  bool dup = false;
  for (int j = 0; j < 31; ++j) { int cj = __shfl(c, j); if (j < k && cj == c) dup = true; }
  if (!dup) atomicAdd(&ssum[c], expf(s - fdec(maxkey[c])));
}

__global__ void lse_kernel(const unsigned* __restrict__ maxkey, const float* __restrict__ ssum,
                           float* __restrict__ lse)
{
  int c = blockIdx.x * 256 + threadIdx.x;
  if (c < C_) lse[c] = fdec(maxkey[c]) + logf(ssum[c]);
}

// obs[n,t,k] = D[text[n,t],k] - lse[w2s[text[n,t],k]]
__global__ void obs_kernel(const int* __restrict__ text, const int* __restrict__ w2s,
                           const float* __restrict__ D, const float* __restrict__ lse,
                           float* __restrict__ obs)
{
  int idx = blockIdx.x * 256 + threadIdx.x;
  if (idx >= N_ * T_ * SPW_) return;
  int k = idx & 31, nt = idx >> 5;
  int v = text[nt];
  int c = w2s[(size_t)v * SPW_ + k];
  obs[idx] = D[(size_t)v * SPW_ + k] - lse[c];
}

// phiT[t,n,j,i] = trans[ci,cj] + obs[n,t+1,j] + (t==0 ? start[ci]+obs[n,0,i] : 0)
__global__ __launch_bounds__(1024) void phi_kernel(
    const int* __restrict__ text, const int* __restrict__ w2s,
    const float* __restrict__ trans, const float* __restrict__ obs,
    const float* __restrict__ start, float* __restrict__ phi)
{
  int t = blockIdx.x, n = blockIdx.y;
  int j = threadIdx.x >> 5, i = threadIdx.x & 31;
  int vi = text[n * T_ + t], vj = text[n * T_ + t + 1];
  int ci = w2s[(size_t)vi * SPW_ + i];
  int cj = w2s[(size_t)vj * SPW_ + j];
  float val = trans[(size_t)ci * C_ + cj] + obs[(size_t)(n * T_ + t + 1) * SPW_ + j];
  if (t == 0) val += start[ci] + obs[(size_t)(n * T_) * SPW_ + i];
  phi[((((size_t)t * N_ + n) * SPW_) + j) * SPW_ + i] = val;
}

// Forward scan (reduction over i = lanes, wave32)
__global__ __launch_bounds__(1024) void fwd_kernel(
    const float* __restrict__ phi, float* __restrict__ alphas_pre,
    float* __restrict__ alphaT, float* __restrict__ logZ, float* __restrict__ outls)
{
  int n = blockIdx.x;
  __shared__ float a[SPW_];
  int j = threadIdx.x >> 5, i = threadIdx.x & 31;
  if (threadIdx.x < SPW_) a[threadIdx.x] = 0.f;
  __syncthreads();
  for (int t = 0; t < T_ - 1; ++t) {
    if (i == 0) alphas_pre[((size_t)t * N_ + n) * SPW_ + j] = a[j];
    float v = a[i] + phi[((((size_t)t * N_ + n) * SPW_) + j) * SPW_ + i];
    float m = v;
    for (int o = 16; o; o >>= 1) m = fmaxf(m, __shfl_xor(m, o));
    float e = expf(v - m);
    for (int o = 16; o; o >>= 1) e += __shfl_xor(e, o);
    float nv = m + logf(e);
    __syncthreads();
    if (i == 0) a[j] = nv;
    __syncthreads();
  }
  if (threadIdx.x < SPW_) alphaT[n * SPW_ + threadIdx.x] = a[threadIdx.x];
  if (j == 0) {
    float av = a[i];
    float m = av;
    for (int o = 16; o; o >>= 1) m = fmaxf(m, __shfl_xor(m, o));
    float e = expf(av - m);
    for (int o = 16; o; o >>= 1) e += __shfl_xor(e, o);
    float lz = m + logf(e);
    if (i == 0) logZ[n] = lz;
    outls[n * SPW_ + i] = av - lz;
  }
}

// Backward scan (reduction over j = lanes)
__global__ __launch_bounds__(1024) void bwd_kernel(
    const float* __restrict__ phi, float* __restrict__ betas)
{
  int n = blockIdx.x;
  __shared__ float b[SPW_];
  int i = threadIdx.x >> 5, j = threadIdx.x & 31;
  if (threadIdx.x < SPW_) b[threadIdx.x] = 0.f;
  __syncthreads();
  for (int t = T_ - 2; t >= 0; --t) {
    if (j == 0) betas[((size_t)t * N_ + n) * SPW_ + i] = b[i];
    float v = phi[((((size_t)t * N_ + n) * SPW_) + j) * SPW_ + i] + b[j];
    float m = v;
    for (int o = 16; o; o >>= 1) m = fmaxf(m, __shfl_xor(m, o));
    float e = expf(v - m);
    for (int o = 16; o; o >>= 1) e += __shfl_xor(e, o);
    float nv = m + logf(e);
    __syncthreads();
    if (j == 0) b[i] = nv;
    __syncthreads();
  }
}

// elbo partials: sum exp(alpha_pre[i] + phi + beta[j] - logZ) * phi per (t,n)
__global__ __launch_bounds__(1024) void elbo_kernel(
    const float* __restrict__ phi, const float* __restrict__ alphas_pre,
    const float* __restrict__ betas, const float* __restrict__ logZ,
    float* __restrict__ partial)
{
  int t = blockIdx.x, n = blockIdx.y;
  int j = threadIdx.x >> 5, i = threadIdx.x & 31;
  float p = phi[((((size_t)t * N_ + n) * SPW_) + j) * SPW_ + i];
  float lm = alphas_pre[((size_t)t * N_ + n) * SPW_ + i] + p +
             betas[((size_t)t * N_ + n) * SPW_ + j] - logZ[n];
  float term = expf(lm) * p;
  for (int o = 16; o; o >>= 1) term += __shfl_xor(term, o);
  __shared__ float wsum[32];
  if (i == 0) wsum[j] = term;
  __syncthreads();
  if (threadIdx.x < 32) {
    float x = wsum[threadIdx.x];
    for (int o = 16; o; o >>= 1) x += __shfl_xor(x, o);
    if (threadIdx.x == 0) partial[(size_t)t * N_ + n] = x;
  }
}

// deterministic final reductions: d_out[0]=elbo, d_out[1]=evidence
__global__ __launch_bounds__(256) void final_kernel(
    const float* __restrict__ partial, int np, const float* __restrict__ logZ, int nz,
    float* __restrict__ d_out)
{
  __shared__ float red[256];
  int tid = threadIdx.x;
  float s = 0.f;
  for (int idx = tid; idx < np; idx += 256) s += partial[idx];
  red[tid] = s; __syncthreads();
  for (int st = 128; st; st >>= 1) { if (tid < st) red[tid] += red[tid + st]; __syncthreads(); }
  if (tid == 0) d_out[0] = red[0];
  __syncthreads();
  float z = 0.f;
  for (int idx = tid; idx < nz; idx += 256) z += logZ[idx];
  red[tid] = z; __syncthreads();
  for (int st = 128; st; st >>= 1) { if (tid < st) red[tid] += red[tid + st]; __syncthreads(); }
  if (tid == 0) d_out[1] = red[0];
}

extern "C" void kernel_launch(void* const* d_in, const int* in_sizes, int n_in,
                              void* d_out, int out_size, void* d_ws, size_t ws_size,
                              hipStream_t stream) {
  (void)in_sizes; (void)n_in; (void)out_size; (void)ws_size;
  const float* start_emb = (const float*)d_in[0];
  const float* sw1 = (const float*)d_in[1];
  const float* sb1 = (const float*)d_in[2];
  const float* sw2 = (const float*)d_in[3];
  const float* sb2 = (const float*)d_in[4];
  const float* s_out_w = (const float*)d_in[5];
  const float* s_out_b = (const float*)d_in[6];
  const float* state_emb = (const float*)d_in[7];
  const float* tw1 = (const float*)d_in[8];
  const float* tb1 = (const float*)d_in[9];
  const float* tw2 = (const float*)d_in[10];
  const float* tb2 = (const float*)d_in[11];
  const float* nsp = (const float*)d_in[12];
  const float* pre_emb = (const float*)d_in[13];
  const float* ew1 = (const float*)d_in[14];
  const float* eb1 = (const float*)d_in[15];
  const float* ew2 = (const float*)d_in[16];
  const float* eb2 = (const float*)d_in[17];
  const float* e_out_w = (const float*)d_in[18];
  const float* e_out_b = (const float*)d_in[19];
  const int* text = (const int*)d_in[20];
  const int* w2s  = (const int*)d_in[21];
  float* out = (float*)d_out;

  // workspace layout (float units)
  float* ws = (float*)d_ws;
  size_t off = 0;
  float* Ys = ws + off;    off += (size_t)C_ * H_;
  float* Yt = ws + off;    off += (size_t)C_ * H_;
  float* Ye = ws + off;    off += (size_t)C_ * H_;
  float* trans = ws + off; off += (size_t)C_ * C_;
  float* Wt = ws + off;    off += (size_t)V_ * H_;
  float* D = ws + off;     off += (size_t)V_ * SPW_;
  unsigned* maxkey = (unsigned*)(ws + off); off += C_;
  float* ssum = ws + off;  off += C_;
  float* lse = ws + off;   off += C_;
  float* slog = ws + off;  off += C_;           // becomes `start` after log_softmax
  float* obs = ws + off;   off += (size_t)N_ * T_ * SPW_;
  float* phi = ws + off;   off += (size_t)(T_ - 1) * N_ * SPW_ * SPW_;
  float* alphas = ws + off; off += (size_t)(T_ - 1) * N_ * SPW_;
  float* betas = ws + off;  off += (size_t)(T_ - 1) * N_ * SPW_;
  float* alphaT = ws + off; off += (size_t)N_ * SPW_;
  float* logZ = ws + off;   off += N_;
  float* partial = ws + off; off += (size_t)(T_ - 1) * N_;
  // f16 staging buffers (sizes in float units: halves/2)
  _Float16* sW1h = (_Float16*)(ws + off); off += (size_t)H_ * H_ / 2;
  _Float16* sW2h = (_Float16*)(ws + off); off += (size_t)H_ * H_ / 2;
  _Float16* tW1h = (_Float16*)(ws + off); off += (size_t)H_ * H_ / 2;
  _Float16* tW2h = (_Float16*)(ws + off); off += (size_t)H_ * H_ / 2;
  _Float16* eW1h = (_Float16*)(ws + off); off += (size_t)H_ * H_ / 2;
  _Float16* eW2h = (_Float16*)(ws + off); off += (size_t)H_ * H_ / 2;
  _Float16* nsph = (_Float16*)(ws + off); off += (size_t)C_ * H_ / 2;

  // 0. stage weights as f16 (n-major for the MLP B operands)
  dim3 tgrid(H_ / 32, H_ / 32), tblk(32, 8);
  wtrans_f16_kernel<<<tgrid, tblk, 0, stream>>>(sw1, sW1h);
  wtrans_f16_kernel<<<tgrid, tblk, 0, stream>>>(sw2, sW2h);
  wtrans_f16_kernel<<<tgrid, tblk, 0, stream>>>(tw1, tW1h);
  wtrans_f16_kernel<<<tgrid, tblk, 0, stream>>>(tw2, tW2h);
  wtrans_f16_kernel<<<tgrid, tblk, 0, stream>>>(ew1, eW1h);
  wtrans_f16_kernel<<<tgrid, tblk, 0, stream>>>(ew2, eW2h);
  cvt_f16_kernel<<<(C_ * H_) / 256, 256, 0, stream>>>(nsp, nsph, C_ * H_);

  // 1. residual MLPs (WMMA)
  mlp_residual_kernel<<<C_ / 16, 512, 0, stream>>>(start_emb, sW1h, sb1, sW2h, sb2, Ys);
  mlp_residual_kernel<<<C_ / 16, 512, 0, stream>>>(state_emb, tW1h, tb1, tW2h, tb2, Yt);
  mlp_residual_kernel<<<C_ / 16, 512, 0, stream>>>(pre_emb, eW1h, eb1, eW2h, eb2, Ye);

  // 2. start distribution
  sout_kernel<<<C_ / 8, 256, 0, stream>>>(Ys, s_out_w, s_out_b, slog);
  row_logsoftmax_kernel<<<1, 256, 0, stream>>>(slog, C_);

  // 3. transition (WMMA GEMM + row log-softmax; double log_softmax == single)
  trans_gemm_kernel<<<dim3(C_ / 16, C_ / 256), 512, 0, stream>>>(Yt, nsph, trans);
  row_logsoftmax_kernel<<<C_, 256, 0, stream>>>(trans, C_);

  // 4. sparse emission log-softmax (only masked entries exist)
  transpose_kernel<<<dim3(V_ / 32, H_ / 32), dim3(32, 8), 0, stream>>>(e_out_w, Wt);
  init_kernel<<<C_ / 256, 256, 0, stream>>>(maxkey, ssum);
  emis_dot_kernel<<<V_ / 8, 256, 0, stream>>>(Ye, Wt, e_out_b, w2s, D, maxkey);
  emis_sum_kernel<<<V_ / 8, 256, 0, stream>>>(D, w2s, maxkey, ssum);
  lse_kernel<<<C_ / 256, 256, 0, stream>>>(maxkey, ssum, lse);
  obs_kernel<<<(N_ * T_ * SPW_) / 256, 256, 0, stream>>>(text, w2s, D, lse, obs);

  // 5. build phiT and run scans
  phi_kernel<<<dim3(T_ - 1, N_), 1024, 0, stream>>>(text, w2s, trans, obs, slog, phi);
  fwd_kernel<<<N_, 1024, 0, stream>>>(phi, alphas, alphaT, logZ, out + 2);
  bwd_kernel<<<N_, 1024, 0, stream>>>(phi, betas);

  // 6. elbo + evidence
  elbo_kernel<<<dim3(T_ - 1, N_), 1024, 0, stream>>>(phi, alphas, betas, logZ, partial);
  final_kernel<<<1, 256, 0, stream>>>(partial, (T_ - 1) * N_, logZ, N_, out);
}
